// Attention_VLM_20959440404935
// MI455X (gfx1250) — matrix operand order repaired and checked by reading, not compiled, not run on hardware
//
#include <hip/hip_runtime.h>
#include <hip/hip_bf16.h>

#define BATCH 2
#define TSEQ  2048
#define DMODEL 2048
#define NHEADS 16
#define NKV    8
#define HDIM   128
#define QSCALE 0.08838834764831845f
#define SOFTCAP 50.0f
#define WINDOW 1024
#define KMASK  -2.3819763e+38f

typedef __bf16 bf16_t;
typedef __attribute__((ext_vector_type(16))) __bf16 v16bf;
typedef __attribute__((ext_vector_type(8)))  __bf16 v8bf;
typedef __attribute__((ext_vector_type(8)))  float  v8f;

// ---------------------------------------------------------------------------
// WMMA wrapper: D = A(16x32 bf16) * B(32x16 bf16) + C(16x16 f32)
// ---------------------------------------------------------------------------
__device__ __forceinline__ v8f wmma_bf16(v16bf a, v16bf b, v8f c) {
  return __builtin_amdgcn_wmma_f32_16x16x32_bf16(
      false, a, false, b, (short)0, c, false, false);
}

// A-fragment (16x32, row-major source, leading dim ld):
// lane row = lane%16; elems 0..7 -> K = 8*(lane/16)+e ; elems 8..15 -> K = 16+8*(lane/16)+e
__device__ __forceinline__ v16bf load_a_frag(const bf16_t* __restrict__ base,
                                             int ld, int row, int k0, int lane) {
  const int hh = lane >> 4;
  const bf16_t* p = base + (size_t)row * ld + k0 + hh * 8;
  union { v16bf v; v8bf h[2]; } u;
  u.h[0] = *reinterpret_cast<const v8bf*>(p);
  u.h[1] = *reinterpret_cast<const v8bf*>(p + 16);
  return u.v;
}

// B-fragment read from a per-lane row with K contiguous in memory:
// col handled by caller (rowptr is per-lane); elems 0..7 -> K = 8*(lane/16)+e ; elems 8..15 -> K = 16+8*(lane/16)+e
__device__ __forceinline__ v16bf load_b_frag_contig(const bf16_t* __restrict__ rowptr,
                                                    int lane) {
  const int hh = lane >> 4;
  const bf16_t* p = rowptr + hh * 8;
  union { v16bf v; v8bf h[2]; } u;
  u.h[0] = *reinterpret_cast<const v8bf*>(p);
  u.h[1] = *reinterpret_cast<const v8bf*>(p + 16);
  return u.v;
}

// ---------------------------------------------------------------------------
// 1a) x: fp32 -> bf16 (same layout)
// ---------------------------------------------------------------------------
__global__ void k_cvt_bf16(const float* __restrict__ src, bf16_t* __restrict__ dst, int n) {
  int i = blockIdx.x * blockDim.x + threadIdx.x;
  if (i < n) dst[i] = (bf16_t)src[i];
}

// 1b) pack wq/wkv into fused, transposed wT[4096][D] (bf16).
__global__ void k_pack_wqkv(const float* __restrict__ wq, const float* __restrict__ wkv,
                            bf16_t* __restrict__ wT) {
  int idx = blockIdx.x * blockDim.x + threadIdx.x;
  if (idx >= 4096 * DMODEL) return;
  const int d  = idx & (DMODEL - 1);
  const int cg = idx >> 11;
  float v;
  if (cg < 2048) {
    const int head = cg >> 7, h = cg & 127;
    v = wq[((size_t)head * DMODEL + d) * HDIM + h];
  } else if (cg < 3072) {
    const int kh = (cg - 2048) >> 7, h = cg & 127;
    v = wkv[((size_t)kh * DMODEL + d) * HDIM + h];
  } else {
    const int kh = (cg - 3072) >> 7, h = cg & 127;
    v = wkv[((size_t)(NKV + kh) * DMODEL + d) * HDIM + h];
  }
  wT[(size_t)cg * DMODEL + d] = (bf16_t)v;
}

// 1c) transpose wo[(n*H+h)][D] -> woT[D][2048] (bf16)
__global__ void k_pack_wo(const float* __restrict__ wo, bf16_t* __restrict__ woT) {
  int idx = blockIdx.x * blockDim.x + threadIdx.x;
  if (idx >= DMODEL * 2048) return;
  const int k = idx & 2047;
  const int d = idx >> 11;
  woT[(size_t)d * 2048 + k] = (bf16_t)wo[(size_t)k * DMODEL + d];
}

// ---------------------------------------------------------------------------
// 2) QKV projection. One wave: 32 tokens x 64 output columns (2x4 WMMA tiles).
//    Outputs transposed: qfT[(b*2048+cg)][T] f32, kfT[(b*1024+c)][T] f32,
//    vT[(b*1024+c)][T] bf16 -> contiguous stores along tokens.
// ---------------------------------------------------------------------------
__global__ void k_qkv_gemm(const bf16_t* __restrict__ xb, const bf16_t* __restrict__ wT,
                           float* __restrict__ qfT, float* __restrict__ kfT,
                           bf16_t* __restrict__ vT) {
  const int lane = threadIdx.x & 31;
  const int wid  = blockIdx.x * (blockDim.x >> 5) + (threadIdx.x >> 5);
  const int ct   = wid & 63;           // 64 column-tiles of width 64
  const int mt   = (wid >> 6) & 63;    // T/32 token tiles
  const int b    = wid >> 12;
  const int col0 = ct * 64;
  const int row0 = mt * 32;
  const int q = lane & 15, hh = lane >> 4;

  const bf16_t* A = xb + (size_t)b * TSEQ * DMODEL;
  const bf16_t* Bp[4];
#pragma unroll
  for (int s = 0; s < 4; ++s) Bp[s] = wT + (size_t)(col0 + s * 16 + q) * DMODEL;

  v8f acc[2][4] = {};
  for (int k0 = 0; k0 < DMODEL; k0 += 32) {
    v16bf a0 = load_a_frag(A, DMODEL, row0 + q, k0, lane);
    v16bf a1 = load_a_frag(A, DMODEL, row0 + 16 + q, k0, lane);
    v16bf bb[4];
#pragma unroll
    for (int s = 0; s < 4; ++s) bb[s] = load_b_frag_contig(Bp[s] + k0, lane);
#pragma unroll
    for (int s = 0; s < 4; ++s) {
      acc[0][s] = wmma_bf16(a0, bb[s], acc[0][s]);
      acc[1][s] = wmma_bf16(a1, bb[s], acc[1][s]);
    }
  }

#pragma unroll
  for (int mi = 0; mi < 2; ++mi) {
    const size_t toff = (size_t)row0 + mi * 16 + 8 * hh;   // 8 contiguous tokens
#pragma unroll
    for (int s = 0; s < 4; ++s) {
      const int cg = col0 + s * 16 + q;
      const v8f& a = acc[mi][s];
      if (cg < 2048) {
        float* p = qfT + ((size_t)b * 2048 + cg) * TSEQ + toff;
        *reinterpret_cast<float4*>(p)     = make_float4(a[0], a[1], a[2], a[3]);
        *reinterpret_cast<float4*>(p + 4) = make_float4(a[4], a[5], a[6], a[7]);
      } else if (cg < 3072) {
        float* p = kfT + ((size_t)b * 1024 + (cg - 2048)) * TSEQ + toff;
        *reinterpret_cast<float4*>(p)     = make_float4(a[0], a[1], a[2], a[3]);
        *reinterpret_cast<float4*>(p + 4) = make_float4(a[4], a[5], a[6], a[7]);
      } else {
        bf16_t* p = vT + ((size_t)b * 1024 + (cg - 3072)) * TSEQ + toff;
        union { uint4 u; bf16_t h[8]; } pk;
#pragma unroll
        for (int r = 0; r < 8; ++r) pk.h[r] = (bf16_t)a[r];
        *reinterpret_cast<uint4*>(p) = pk.u;
      }
    }
  }
}

// ---------------------------------------------------------------------------
// 3) RoPE (+ query scale): read transposed fp32, write [B,T,heads,H] bf16
// ---------------------------------------------------------------------------
__global__ void k_rope(const float* __restrict__ qfT, const float* __restrict__ kfT,
                       const int* __restrict__ segpos,
                       bf16_t* __restrict__ qb, bf16_t* __restrict__ kb) {
  const int QP = BATCH * TSEQ * NHEADS * (HDIM / 2);
  const int KP = BATCH * TSEQ * NKV * (HDIM / 2);
  int idx = blockIdx.x * blockDim.x + threadIdx.x;
  if (idx >= QP + KP) return;
  const bool isq = idx < QP;
  const int i = isq ? idx : idx - QP;
  const int heads = isq ? NHEADS : NKV;
  const int h  = i % (HDIM / 2);
  const int hd = (i / (HDIM / 2)) % heads;
  const int t  = (i / ((HDIM / 2) * heads)) % TSEQ;
  const int b  = i / ((HDIM / 2) * heads * TSEQ);

  const float pos  = (float)segpos[b * TSEQ + t];
  const float freq = __expf(-(2.0f * h / (float)HDIM) * 9.210340371976184f); // ln(1e4)
  float sv, cv;
  __sincosf(pos * freq, &sv, &cv);

  const float* src = isq ? qfT : kfT;
  const size_t srow = ((size_t)b * heads + hd) * HDIM + h;
  const float first  = src[srow * TSEQ + t];
  const float second = src[(srow + HDIM / 2) * TSEQ + t];
  const float scale  = isq ? QSCALE : 1.0f;

  bf16_t* dst = isq ? qb : kb;
  const size_t off = (((size_t)b * TSEQ + t) * heads + hd) * HDIM;
  dst[off + h]            = (bf16_t)((first * cv - second * sv) * scale);
  dst[off + h + HDIM / 2] = (bf16_t)((second * cv + first * sv) * scale);
}

// ---------------------------------------------------------------------------
// 4) Flash attention. One wave: 16 queries x one head; S^T = K*Q^T, enc^T = V^T*P^T.
// ---------------------------------------------------------------------------
#define ATTN_WAVES 4
__global__ void k_attn(const bf16_t* __restrict__ qb, const bf16_t* __restrict__ kb,
                       const bf16_t* __restrict__ vT, bf16_t* __restrict__ enc) {
  __shared__ __align__(16) bf16_t ldsP[ATTN_WAVES][16 * 32];
  const int lane = threadIdx.x & 31;
  const int slot = threadIdx.x >> 5;
  const int wid  = blockIdx.x * ATTN_WAVES + slot;
  const int qt = wid & 127;
  const int n  = (wid >> 7) & (NHEADS - 1);
  const int b  = wid >> 11;
  const int kh = n >> 1;                   // G = 2
  const int t0 = qt * 16;
  const int hh = lane >> 4, q = lane & 15;
  const int tq = t0 + q;

  const bf16_t* qrow = qb + (((size_t)b * TSEQ + tq) * NHEADS + n) * HDIM;
  v16bf qfrag[4];
#pragma unroll
  for (int c = 0; c < 4; ++c) qfrag[c] = load_b_frag_contig(qrow + c * 32, lane);

  v8f acc[8] = {};
  float m = -1e30f, l = 0.0f;

  const int sstride = NKV * HDIM;
  const bf16_t* Kbase = kb + ((size_t)b * TSEQ * NKV + kh) * HDIM;
  const bf16_t* VTb   = vT + ((size_t)b * NKV + kh) * HDIM * TSEQ;

  int s_begin = t0 - (WINDOW - 1);
  if (s_begin < 0) s_begin = 0;
  s_begin &= ~31;

  for (int s0 = s_begin; s0 <= t0 + 15; s0 += 32) {
    // prefetch next tile's K rows and V columns into cache (global_prefetch_b8)
    if (s0 + 32 <= t0 + 15) {
      __builtin_prefetch(Kbase + (size_t)(s0 + 32 + q) * sstride, 0, 3);
      __builtin_prefetch(VTb + (size_t)(q * 16) * TSEQ + s0 + 32, 0, 3);
    }

    v8f c0 = {}, c1 = {};
#pragma unroll
    for (int c = 0; c < 4; ++c) {
      v16bf a0 = load_a_frag(Kbase + (size_t)s0 * sstride, sstride, q, c * 32, lane);
      c0 = wmma_bf16(a0, qfrag[c], c0);
      v16bf a1 = load_a_frag(Kbase + (size_t)(s0 + 16) * sstride, sstride, q, c * 32, lane);
      c1 = wmma_bf16(a1, qfrag[c], c1);
    }

    float pv[16];
    float tmax = -1e30f;
#pragma unroll
    for (int j = 0; j < 2; ++j) {
      v8f cc = j ? c1 : c0;
#pragma unroll
      for (int r = 0; r < 8; ++r) {
        const int s = s0 + j * 16 + 8 * hh + r;
        float v = cc[r];
        v = SOFTCAP * tanhf(v * (1.0f / SOFTCAP));
        const bool ok = (s <= tq) && (s > tq - WINDOW);
        v = ok ? v : KMASK;
        pv[j * 8 + r] = v;
        tmax = fmaxf(tmax, v);
      }
    }
    tmax = fmaxf(tmax, __shfl_xor(tmax, 16, 32));
    const float mnew  = fmaxf(m, tmax);
    const float scale = __expf(m - mnew);

    float lsum = 0.0f;
    union { uint4 u; bf16_t h[8]; } pk[2];
#pragma unroll
    for (int j = 0; j < 2; ++j) {
#pragma unroll
      for (int r = 0; r < 8; ++r) {
        const float p = __expf(pv[j * 8 + r] - mnew);
        lsum += p;
        pk[j].h[r] = (bf16_t)p;
      }
    }
    lsum += __shfl_xor(lsum, 16, 32);
    l = l * scale + lsum;
    m = mnew;
#pragma unroll
    for (int i = 0; i < 8; ++i)
#pragma unroll
      for (int r = 0; r < 8; ++r) acc[i][r] *= scale;

    *reinterpret_cast<uint4*>(&ldsP[slot][q * 32 + 8 * hh])      = pk[0].u;
    *reinterpret_cast<uint4*>(&ldsP[slot][q * 32 + 16 + 8 * hh]) = pk[1].u;
    v16bf pB = load_b_frag_contig(&ldsP[slot][q * 32], lane);

#pragma unroll
    for (int i = 0; i < 8; ++i) {
      v16bf vfrag = load_a_frag(VTb, TSEQ, i * 16 + q, s0, lane);
      acc[i] = wmma_bf16(vfrag, pB, acc[i]);
    }
  }

  const float inv = 1.0f / l;
  bf16_t* erow = enc + (((size_t)b * TSEQ + tq) * NHEADS + n) * HDIM;
#pragma unroll
  for (int i = 0; i < 8; ++i) {
    union { uint4 u; bf16_t h[8]; } o;
#pragma unroll
    for (int r = 0; r < 8; ++r) o.h[r] = (bf16_t)(acc[i][r] * inv);
    *reinterpret_cast<uint4*>(erow + i * 16 + 8 * hh) = o.u;
  }
}

// ---------------------------------------------------------------------------
// 5) Output projection, transposed: out^T = wo^T * enc^T.
//    One wave: 32 d-rows x 64 tokens (2x4 WMMA tiles), contiguous float4 stores.
// ---------------------------------------------------------------------------
__global__ void k_out_gemm(const bf16_t* __restrict__ enc,
                           const bf16_t* __restrict__ woT,
                           float* __restrict__ out) {
  const int lane = threadIdx.x & 31;
  const int wid  = blockIdx.x * (blockDim.x >> 5) + (threadIdx.x >> 5);
  const int ct = wid & 63;                 // D/32 d-tiles
  const int mt = (wid >> 6) & 31;          // T/64 token tiles
  const int b  = wid >> 11;
  const int d0 = ct * 32;
  const int t0 = mt * 64;
  const int KD = NHEADS * HDIM;            // 2048
  const int q = lane & 15, hh = lane >> 4;

  const bf16_t* Brow[4];
#pragma unroll
  for (int j = 0; j < 4; ++j)
    Brow[j] = enc + ((size_t)b * TSEQ + t0 + j * 16 + q) * KD;

  v8f acc[2][4] = {};
  for (int k0 = 0; k0 < KD; k0 += 32) {
    v16bf a0 = load_a_frag(woT, KD, d0 + q, k0, lane);
    v16bf a1 = load_a_frag(woT, KD, d0 + 16 + q, k0, lane);
    v16bf bb[4];
#pragma unroll
    for (int j = 0; j < 4; ++j) bb[j] = load_b_frag_contig(Brow[j] + k0, lane);
#pragma unroll
    for (int j = 0; j < 4; ++j) {
      acc[0][j] = wmma_bf16(a0, bb[j], acc[0][j]);
      acc[1][j] = wmma_bf16(a1, bb[j], acc[1][j]);
    }
  }

#pragma unroll
  for (int mi = 0; mi < 2; ++mi) {
#pragma unroll
    for (int j = 0; j < 4; ++j) {
      const v8f& a = acc[mi][j];
      float* p = out + ((size_t)b * TSEQ + t0 + j * 16 + q) * DMODEL + d0 + mi * 16 + 8 * hh;
      *reinterpret_cast<float4*>(p)     = make_float4(a[0], a[1], a[2], a[3]);
      *reinterpret_cast<float4*>(p + 4) = make_float4(a[4], a[5], a[6], a[7]);
    }
  }
}

// ---------------------------------------------------------------------------
extern "C" void kernel_launch(void* const* d_in, const int* in_sizes, int n_in,
                              void* d_out, int out_size, void* d_ws, size_t ws_size,
                              hipStream_t stream) {
  const float* x    = (const float*)d_in[0];
  const int*   spos = (const int*)d_in[1];
  // d_in[2] = attn_mask (causal tril) -> implemented analytically
  const float* wq   = (const float*)d_in[3];
  const float* wkv  = (const float*)d_in[4];
  const float* wo   = (const float*)d_in[5];
  float* out = (float*)d_out;

  const size_t nX  = (size_t)BATCH * TSEQ * DMODEL;          // 8388608
  const size_t nWT = (size_t)4096 * DMODEL;                  // 8388608
  const size_t nWO = (size_t)DMODEL * 2048;                  // 4194304
  const size_t nQ  = (size_t)BATCH * TSEQ * NHEADS * HDIM;   // 8388608
  const size_t nK  = (size_t)BATCH * TSEQ * NKV * HDIM;      // 4194304

  char* w = (char*)d_ws;
  bf16_t* xb   = (bf16_t*)w;  w += nX * 2;
  bf16_t* wT   = (bf16_t*)w;  w += nWT * 2;
  bf16_t* woT  = (bf16_t*)w;  w += nWO * 2;
  float*  qfT  = (float*)w;   w += nQ * 4;
  float*  kfT  = (float*)w;   w += nK * 4;
  bf16_t* qbuf = (bf16_t*)w;  w += nQ * 2;
  bf16_t* kbuf = (bf16_t*)w;  w += nK * 2;
  bf16_t* vT   = (bf16_t*)w;  w += nK * 2;
  bf16_t* encb = (bf16_t*)qfT;   // qfT dead after rope; reuse for encoded

  // 1) conversions / packing
  k_cvt_bf16 <<<(int)((nX + 255) / 256), 256, 0, stream>>>(x, xb, (int)nX);
  k_pack_wqkv<<<(int)((nWT + 255) / 256), 256, 0, stream>>>(wq, wkv, wT);
  k_pack_wo  <<<(int)((nWO + 255) / 256), 256, 0, stream>>>(wo, woT);

  // 2) QKV projection: 8192 waves (32x64 tiles), 8 waves/block
  k_qkv_gemm<<<1024, 256, 0, stream>>>(xb, wT, qfT, kfT, vT);

  // 3) RoPE
  {
    const int total = BATCH * TSEQ * (NHEADS + NKV) * (HDIM / 2);
    k_rope<<<(total + 255) / 256, 256, 0, stream>>>(qfT, kfT, spos, qbuf, kbuf);
  }

  // 4) attention: 4096 waves, 4 waves/block
  k_attn<<<1024, 32 * ATTN_WAVES, 0, stream>>>(qbuf, kbuf, vT, encb);

  // 5) output projection: 4096 waves (32x64 tiles), 8 waves/block
  k_out_gemm<<<512, 256, 0, stream>>>(encb, woT, out);
}
